// RiemannFMGeodesicAttention_1245540515918
// MI455X (gfx1250) — matrix operand order repaired and checked by reading, not compiled, not run on hardware
//
#include <hip/hip_runtime.h>

#define DEV_ __device__ __forceinline__

typedef __attribute__((ext_vector_type(16))) __bf16 bf16x16;
typedef __attribute__((ext_vector_type(8)))  __bf16 bf16x8;
typedef __attribute__((ext_vector_type(8)))  float  f32x8;

constexpr int B_  = 4;
constexpr int N_  = 1024;
constexpr int DM_ = 512;
constexpr int H_  = 8;
constexpr int C_  = 3;
constexpr int DK_ = 64;
constexpr int R_  = B_ * N_;   // 4096 flattened (b,n) rows
constexpr int XA_ = 32;        // augmented x row length (bf16)

// ---------------------------------------------------------------------------
// WMMA helper: D = A(16x32 bf16) * B(32x16 bf16) + C(16x16 f32)
// ---------------------------------------------------------------------------
DEV_ f32x8 wmma_bf16(bf16x16 a, bf16x16 b, f32x8 c) {
  return __builtin_amdgcn_wmma_f32_16x16x32_bf16(false, a, false, b,
                                                 (short)0, c, false, false);
}

// A-operand (16x32, 16-bit): lane L holds row M=L%16.
// elems 0..7  -> K = kbase + (hi?8:0) + e
// elems 8..15 -> K = kbase + 16 + (hi?8:0) + (e-8)
DEV_ bf16x16 loadA16(const __bf16* row, int kbase, int hi8) {
  bf16x8 lo = *(const bf16x8*)(row + kbase + hi8);
  bf16x8 up = *(const bf16x8*)(row + kbase + 16 + hi8);
  bf16x16 r;
#pragma unroll
  for (int e = 0; e < 8; ++e) { r[e] = lo[e]; r[8 + e] = up[e]; }
  return r;
}

// B-operand (32x16, 16-bit): lane L holds col N=L%16.
// elems 0..7 -> K = (hi?8:0) + e ; elems 8..15 -> K = 16 + (hi?8:0) + (e-8); p points at K=(hi?8:0).
DEV_ bf16x16 loadB16(const __bf16* p) {
  bf16x8 lo = *(const bf16x8*)(p);
  bf16x8 up = *(const bf16x8*)(p + 16);
  bf16x16 r;
#pragma unroll
  for (int e = 0; e < 8; ++e) { r[e] = lo[e]; r[8 + e] = up[e]; }
  return r;
}

// f32 -> bf16 on-the-fly operand builders (for the projection GEMMs)
DEV_ bf16x16 cvtA(const float* row, int kbase, int hi8) {
  const float* p0 = row + kbase + hi8;
  const float* p1 = p0 + 16;
  bf16x16 r;
#pragma unroll
  for (int e = 0; e < 8; ++e) { r[e] = (__bf16)p0[e]; r[8 + e] = (__bf16)p1[e]; }
  return r;
}
DEV_ bf16x16 cvtB(const float* row, int kbase, int hi16) {
  const float* p = row + kbase + (hi16 >> 1);
  bf16x16 r;
#pragma unroll
  for (int e = 0; e < 16; ++e) r[e] = (__bf16)p[e < 8 ? e : e + 8];
  return r;
}

// ---------------------------------------------------------------------------
// Kernel: build augmented x (bf16): K-side [x, -0.5|x|^2, 1, 0...],
//                                   Q-side [x, 1, -0.5|x|^2, 0...]
// so that WMMA dot = x_i.x_j - 0.5|x_i|^2 - 0.5|x_j|^2 = -d^2/2.
// ---------------------------------------------------------------------------
__global__ __launch_bounds__(256)
void prep_x_kernel(const float* __restrict__ x,
                   __bf16* __restrict__ xaK, __bf16* __restrict__ xaQ) {
  const int t = blockIdx.x * 256 + threadIdx.x;   // (b*N + n)*C + c
  if (t >= B_ * N_ * C_) return;
  const float* xp = x + (size_t)t * 16;
  __bf16* kq = xaK + (size_t)t * XA_;
  __bf16* qq = xaQ + (size_t)t * XA_;
  float sq = 0.f;
#pragma unroll
  for (int d = 0; d < 16; ++d) {
    const float v = xp[d];
    sq += v * v;
    kq[d] = (__bf16)v;
    qq[d] = (__bf16)v;
  }
  const __bf16 mh = (__bf16)(-0.5f * sq);
  const __bf16 one = (__bf16)1.0f;
  const __bf16 zero = (__bf16)0.0f;
  kq[16] = mh;  kq[17] = one;
  qq[16] = one; qq[17] = mh;
#pragma unroll
  for (int d = 18; d < XA_; ++d) { kq[d] = zero; qq[d] = zero; }
}

// ---------------------------------------------------------------------------
// WMMA GEMM: out[n,m] = sum_k A[n,k] * W[m,k] + bias[m]   (A: R_ x 512)
// MODE 0: store bf16 per-head (B,H,N,DK)   [Q, K]
// MODE 1: store bf16 transposed (B,H,DK,N) [V]
// MODE 2: store f32 (R_, DM_) to d_out     [final projection]
// One wave computes one 16x16 output tile; 8 waves / block.
// ---------------------------------------------------------------------------
template <int MODE>
__global__ __launch_bounds__(256)
void gemm_wmma_kernel(const float* __restrict__ A, const float* __restrict__ W,
                      const float* __restrict__ bias,
                      __bf16* __restrict__ dbf, float* __restrict__ df32) {
  const int lane = threadIdx.x & 31;
  const int wave = threadIdx.x >> 5;
  const int li = lane & 15, hi = lane >> 4;
  const int hi8 = hi * 8, hi16 = hi * 16;

  const int wid = blockIdx.x * 8 + wave;   // 8192 tiles = 256 x 32
  const int tm = wid & 31;
  const int tn = wid >> 5;
  const int n0 = tn * 16, m0 = tm * 16;

  const float* arow = A + (size_t)(n0 + li) * DM_;
  const float* wrow = W + (size_t)(m0 + li) * DM_;

  f32x8 acc = {};
  for (int k0 = 0; k0 < DM_; k0 += 32) {
    bf16x16 a = cvtA(arow, k0, hi8);
    bf16x16 b = cvtB(wrow, k0, hi16);
    acc = wmma_bf16(a, b, acc);
  }

  const int m = m0 + li;
  const float bi = bias[m];
#pragma unroll
  for (int r = 0; r < 8; ++r) {
    const float v = acc[r] + bi;
    const int gn = n0 + r + hi8;             // flattened (b,n)
    if (MODE == 2) {
      df32[(size_t)gn * DM_ + m] = v;
    } else {
      const int bb = gn >> 10, n = gn & (N_ - 1);
      const int head = m >> 6, d = m & (DK_ - 1);
      if (MODE == 0)
        dbf[(((size_t)(bb * H_ + head)) * N_ + n) * DK_ + d] = (__bf16)v;
      else
        dbf[(((size_t)(bb * H_ + head)) * DK_ + d) * N_ + n] = (__bf16)v;
    }
  }
}

// ---------------------------------------------------------------------------
// Fused flash attention with geodesic kernel.
// One wave owns a 16-row query tile of one (b,h); loops j in chunks of 32.
// Computes S^T = K.Q^T so softmax state is per-lane and P^T repacks directly
// into the A-operand of the P.V WMMA.
// ---------------------------------------------------------------------------
__global__ __launch_bounds__(256)
void attn_kernel(const __bf16* __restrict__ Qb, const __bf16* __restrict__ Kb,
                 const __bf16* __restrict__ Vt, const __bf16* __restrict__ xaK,
                 const __bf16* __restrict__ xaQ,
                 const float* __restrict__ alpha,
                 const float* __restrict__ log_sigma,
                 float* __restrict__ Obuf) {
  const int lane = threadIdx.x & 31;
  const int wave = threadIdx.x >> 5;
  const int li = lane & 15, hi = lane >> 4;
  const int hi8 = hi * 8, hi16 = hi * 16;

  const int tile = blockIdx.x * 8 + wave;        // B_*H_*(N_/16) = 2048
  const int TI = N_ / 16;
  const int i0 = (tile % TI) * 16;
  const int bh = tile / TI;
  const int h = bh & (H_ - 1), b = bh >> 3;

  const __bf16* Qp = Qb + (size_t)bh * N_ * DK_;
  const __bf16* Kp = Kb + (size_t)bh * N_ * DK_;
  const __bf16* Vp = Vt + (size_t)bh * DK_ * N_;
  const __bf16* xKp = xaK + (size_t)b * N_ * C_ * XA_;
  const __bf16* xQp = xaQ + (size_t)b * N_ * C_ * XA_;

  float al[C_], iv[C_];
#pragma unroll
  for (int c = 0; c < C_; ++c) {
    al[c] = alpha[h * C_ + c];
    float s = __expf(log_sigma[h * C_ + c]);
    s = fmaxf(s, 1e-4f);
    iv[c] = 1.0f / (s * s);
  }

  // Q^T as B-operand (lane = query column i), two K-steps over dk=64.
  bf16x16 qB[2];
#pragma unroll
  for (int s = 0; s < 2; ++s)
    qB[s] = loadB16(Qp + (size_t)(i0 + li) * DK_ + s * 32 + hi8);
  // augmented x_q^T as B-operand per component
  bf16x16 xqB[C_];
#pragma unroll
  for (int c = 0; c < C_; ++c)
    xqB[c] = loadB16(xQp + ((size_t)(i0 + li) * C_ + c) * XA_ + hi8);

  f32x8 O0 = {}, O1 = {}, O2 = {}, O3 = {};
  float m = -1e30f, l = 0.f;
  const float scale = 0.125f;   // 1/sqrt(64)

  for (int j0 = 0; j0 < N_; j0 += 32) {
    if (j0 + 32 < N_) {
      __builtin_prefetch(Kp + (size_t)(j0 + 32 + li) * DK_, 0, 3);
      __builtin_prefetch(Vp + (size_t)li * N_ + j0 + 32, 0, 3);
    }
    // --- scores S^T = K.Q^T : two 16x16 D tiles (j halves) -----------------
    f32x8 st[2] = {};
#pragma unroll
    for (int jt = 0; jt < 2; ++jt) {
      const __bf16* krow = Kp + (size_t)(j0 + jt * 16 + li) * DK_;
#pragma unroll
      for (int s = 0; s < 2; ++s)
        st[jt] = wmma_bf16(loadA16(krow, s * 32, hi8), qB[s], st[jt]);
    }
    // --- geodesic -d^2/2 per component -------------------------------------
    f32x8 gg[2][C_];
#pragma unroll
    for (int jt = 0; jt < 2; ++jt)
#pragma unroll
      for (int c = 0; c < C_; ++c) {
        f32x8 z = {};
        const __bf16* xr = xKp + ((size_t)(j0 + jt * 16 + li) * C_ + c) * XA_;
        gg[jt][c] = wmma_bf16(loadA16(xr, 0, hi8), xqB[c], z);
      }
    // --- elementwise: score + sum_c alpha*exp(-d^2/(2 sigma^2)) ------------
    float p[16];
    float cmax = -1e30f;
#pragma unroll
    for (int jt = 0; jt < 2; ++jt)
#pragma unroll
      for (int r = 0; r < 8; ++r) {
        float sv = st[jt][r] * scale;
#pragma unroll
        for (int c = 0; c < C_; ++c)
          sv += al[c] * __expf(gg[jt][c][r] * iv[c]);
        p[jt * 8 + r] = sv;
        cmax = fmaxf(cmax, sv);
      }
    cmax = fmaxf(cmax, __shfl_xor(cmax, 16, 32));
    const float mnew = fmaxf(m, cmax);
    const float corr = __expf(m - mnew);
    m = mnew;
    l *= corr;
#pragma unroll
    for (int idx = 0; idx < 16; ++idx) {
      p[idx] = __expf(p[idx] - m);
      l += p[idx];
    }
    // --- rescale O accumulators (row-indexed: broadcast corr across lanes) -
    float cr[8];
#pragma unroll
    for (int r = 0; r < 8; ++r) cr[r] = __shfl(corr, r + hi8, 32);
#pragma unroll
    for (int r = 0; r < 8; ++r) {
      O0[r] *= cr[r]; O1[r] *= cr[r]; O2[r] *= cr[r]; O3[r] *= cr[r];
    }
    // --- pack P^T into A-operand and accumulate O += P.V -------------------
    bf16x16 pa;
#pragma unroll
    for (int e = 0; e < 16; ++e) pa[e] = (__bf16)p[e];
    O0 = wmma_bf16(pa, loadB16(Vp + (size_t)(0 * 16 + li) * N_ + j0 + hi8), O0);
    O1 = wmma_bf16(pa, loadB16(Vp + (size_t)(1 * 16 + li) * N_ + j0 + hi8), O1);
    O2 = wmma_bf16(pa, loadB16(Vp + (size_t)(2 * 16 + li) * N_ + j0 + hi8), O2);
    O3 = wmma_bf16(pa, loadB16(Vp + (size_t)(3 * 16 + li) * N_ + j0 + hi8), O3);
  }

  // --- normalize and store O as f32 (B,N, h*64+d) ---------------------------
  const float lt = l + __shfl_xor(l, 16, 32);
  const float linv = 1.0f / lt;
  float lr[8];
#pragma unroll
  for (int r = 0; r < 8; ++r) lr[r] = __shfl(linv, r + hi8, 32);
#pragma unroll
  for (int r = 0; r < 8; ++r) {
    const size_t rowoff = (size_t)(b * N_ + i0 + r + hi8) * DM_ + h * DK_ + li;
    Obuf[rowoff + 0]  = O0[r] * lr[r];
    Obuf[rowoff + 16] = O1[r] * lr[r];
    Obuf[rowoff + 32] = O2[r] * lr[r];
    Obuf[rowoff + 48] = O3[r] * lr[r];
  }
}

// ---------------------------------------------------------------------------
extern "C" void kernel_launch(void* const* d_in, const int* in_sizes, int n_in,
                              void* d_out, int out_size, void* d_ws, size_t ws_size,
                              hipStream_t stream) {
  const float* h  = (const float*)d_in[0];
  const float* x  = (const float*)d_in[1];
  const float* Wq = (const float*)d_in[2];
  const float* bq = (const float*)d_in[3];
  const float* Wk = (const float*)d_in[4];
  const float* bk = (const float*)d_in[5];
  const float* Wv = (const float*)d_in[6];
  const float* bv = (const float*)d_in[7];
  const float* Wo = (const float*)d_in[8];
  const float* bo = (const float*)d_in[9];
  const float* alpha = (const float*)d_in[10];
  const float* lsig  = (const float*)d_in[11];
  float* out = (float*)d_out;
  (void)in_sizes; (void)n_in; (void)out_size; (void)ws_size;

  char* ws = (char*)d_ws;
  size_t off = 0;
  auto take = [&](size_t bytes) {
    size_t r = off;
    off += (bytes + 255) & ~(size_t)255;
    return r;
  };
  const size_t qkv_bytes = (size_t)B_ * H_ * N_ * DK_ * sizeof(__bf16);
  const size_t xa_bytes  = (size_t)B_ * N_ * C_ * XA_ * sizeof(__bf16);
  __bf16* Qb  = (__bf16*)(ws + take(qkv_bytes));
  __bf16* Kb  = (__bf16*)(ws + take(qkv_bytes));
  __bf16* Vtb = (__bf16*)(ws + take(qkv_bytes));
  __bf16* xK  = (__bf16*)(ws + take(xa_bytes));
  __bf16* xQ  = (__bf16*)(ws + take(xa_bytes));
  float*  Ob  = (float*)(ws + take((size_t)R_ * DM_ * sizeof(float)));

  prep_x_kernel<<<(B_ * N_ * C_ + 255) / 256, 256, 0, stream>>>(x, xK, xQ);

  const int gemm_blocks = (R_ / 16) * (DM_ / 16) / 8;   // 1024
  gemm_wmma_kernel<0><<<gemm_blocks, 256, 0, stream>>>(h, Wq, bq, Qb, nullptr);
  gemm_wmma_kernel<0><<<gemm_blocks, 256, 0, stream>>>(h, Wk, bk, Kb, nullptr);
  gemm_wmma_kernel<1><<<gemm_blocks, 256, 0, stream>>>(h, Wv, bv, Vtb, nullptr);

  const int attn_blocks = B_ * H_ * (N_ / 16) / 8;      // 256
  attn_kernel<<<attn_blocks, 256, 0, stream>>>(Qb, Kb, Vtb, xK, xQ, alpha, lsig, Ob);

  gemm_wmma_kernel<2><<<gemm_blocks, 256, 0, stream>>>(Ob, Wo, bo, nullptr, out);
}
